// FlashOccTemporalMixer_16097537425484
// MI455X (gfx1250) — matrix operand order repaired and checked by reading, not compiled, not run on hardware
//
#include <hip/hip_runtime.h>
#include <hip/hip_bf16.h>

// ---------------------------------------------------------------------------
// FlashOccTemporalMixer for MI455X (gfx1250, wave32, WMMA)
//
//   k_pack_w      : conv_w[o][c][k] f32 -> Wp[k][o][c] bf16 ; proj_w -> Pb bf16
//   k_align_pack  : homography warp + bilinear sample, write X[t][n][c] bf16
//   k_conv1d_wmma : Y[o][t][n] f32 = sum_k Wp[k] @ X[t+k-1]
//                   (async-to-LDS B staging + 2x2 register-blocked v_wmma bf16)
//   k_bn1_gelu    : BatchNorm1d + erf-GELU -> tokens out [n][c][t] f32,
//                   plus bf16 copy of t=7 slice G[n][c] for the proj GEMM
//   k_proj_wmma   : F[o][n] f32 = Pb @ G (same LDS-staged WMMA scheme)
//   k_bn2_gelu    : BatchNorm2d + erf-GELU -> fused out [o][n] f32
// ---------------------------------------------------------------------------

typedef __bf16 bf16_t;
typedef __attribute__((ext_vector_type(16))) __bf16 v16bf;
typedef __attribute__((ext_vector_type(8)))  __bf16 v8bf;
typedef __attribute__((ext_vector_type(8)))  float  v8f;
typedef __attribute__((ext_vector_type(4)))  int    v4i;
typedef __attribute__((address_space(1)))    v4i    as1_v4i;  // global int4
typedef __attribute__((address_space(3)))    v4i    as3_v4i;  // LDS int4

static constexpr int kC    = 256;       // channels
static constexpr int kT    = 8;         // time steps
static constexpr int kHW   = 16384;     // H*W = 128*128
static constexpr int kW    = 128;
static constexpr int kNL   = kHW * kT;  // 131072 samples/channel for BN1
static constexpr int kLP   = 264;       // padded LDS row length (bf16 elems):
                                        // 528B stride -> 4*row mod 64 banks,
                                        // conflict-free 16-lane ds_load_b128

#if __has_builtin(__builtin_amdgcn_global_load_async_to_lds_b128) && \
    __has_builtin(__builtin_amdgcn_s_wait_asynccnt)
#define USE_ASYNC_LDS 1
#endif

// Move 16B global -> LDS (async path uses gfx1250 GLOBAL_LOAD_ASYNC_TO_LDS_B128,
// tracked by ASYNCcnt; fallback bounces through VGPRs).
// Builtin signature (from hipcc diagnostic): (v4i AS1*, v4i AS3*, Imm, Imm).
__device__ __forceinline__ void stage16(const bf16_t* g, bf16_t* l) {
#ifdef USE_ASYNC_LDS
  __builtin_amdgcn_global_load_async_to_lds_b128(
      (as1_v4i*)(uintptr_t)g,
      (as3_v4i*)(unsigned)(uintptr_t)l,   // low 32 bits of flat ptr = LDS offset
      0, 0);
#else
  *(v8bf*)l = *(const v8bf*)g;
#endif
}

__device__ __forceinline__ void stage_fence() {
#ifdef USE_ASYNC_LDS
  __builtin_amdgcn_s_wait_asynccnt(0);
#endif
  __syncthreads();
}

// ---------------------------------------------------------------------------
// Weight repack: conv_w [256][256][3] f32 -> Wp [3][256][256] bf16 (row-major
// over c so A-fragments are contiguous 16B loads); proj_w [256][256] -> bf16.
// ---------------------------------------------------------------------------
__global__ void k_pack_w(const float* __restrict__ conv_w,
                         const float* __restrict__ proj_w,
                         bf16_t* __restrict__ Wp,
                         bf16_t* __restrict__ Pb) {
  int i = blockIdx.x * 256 + threadIdx.x;
  if (i < 3 * kC * kC) {
    int k = i % 3;
    int rem = i / 3;
    int c = rem & 255;
    int o = rem >> 8;
    Wp[(size_t)k * (kC * kC) + (size_t)o * kC + c] = (bf16_t)conv_w[i];
  } else {
    int j = i - 3 * kC * kC;   // j < 65536
    Pb[j] = (bf16_t)proj_w[j];
  }
}

// ---------------------------------------------------------------------------
// Warp history frames to the key frame and pack to bf16 X[t][n][c].
// One block per (w,h,t) pixel; 256 threads = 256 channels. align_corners=True
// makes the sampling coordinate exactly the warped pixel coordinate:
// p = (inv_f2b * inv(H_t) * f2b) * [w, h, 1]^T  (reference drops warped[...,2]).
// ---------------------------------------------------------------------------
__global__ void k_align_pack(const float* __restrict__ bev,
                             const float* __restrict__ h2k,
                             bf16_t* __restrict__ X) {
  const int w = blockIdx.x;
  const int h = blockIdx.y;
  const int t = blockIdx.z;
  const int c = threadIdx.x;
  const int n = h * kW + w;
  const size_t frame = (size_t)t * kC * kHW;

  float v;
  if (t == 0) {
    v = bev[frame + (size_t)c * kHW + n];
  } else {
    const float* M = h2k + t * 9;
    float a00 = M[0], a01 = M[1], a02 = M[2];
    float a10 = M[3], a11 = M[4], a12 = M[5];
    float a20 = M[6], a21 = M[7], a22 = M[8];
    // inverse of H_t via adjugate
    float c00 =  (a11 * a22 - a12 * a21);
    float c01 = -(a10 * a22 - a12 * a20);
    float c02 =  (a10 * a21 - a11 * a20);
    float id  = 1.0f / (a00 * c00 + a01 * c01 + a02 * c02);
    float i00 =  c00 * id;
    float i01 = -(a01 * a22 - a02 * a21) * id;
    float i02 =  (a01 * a12 - a02 * a11) * id;
    float i10 =  c01 * id;
    float i11 =  (a00 * a22 - a02 * a20) * id;
    float i12 = -(a00 * a12 - a02 * a10) * id;
    float i20 =  c02 * id;
    float i21 = -(a00 * a21 - a01 * a20) * id;
    float i22 =  (a00 * a11 - a01 * a10) * id;
    // B1 = invH * f2b ; f2b = [[0.625,0,-40],[0,0.625,-40],[0,0,1]]
    float b00 = i00 * 0.625f, b01 = i01 * 0.625f, b02 = -40.f * (i00 + i01) + i02;
    float b10 = i10 * 0.625f, b11 = i11 * 0.625f, b12 = -40.f * (i10 + i11) + i12;
    float b20 = i20 * 0.625f, b21 = i21 * 0.625f, b22 = -40.f * (i20 + i21) + i22;
    // A = inv_f2b * B1 ; inv_f2b = [[1.6,0,64],[0,1.6,64],[0,0,1]]
    float A00 = 1.6f * b00 + 64.f * b20, A01 = 1.6f * b01 + 64.f * b21, A02 = 1.6f * b02 + 64.f * b22;
    float A10 = 1.6f * b10 + 64.f * b20, A11 = 1.6f * b11 + 64.f * b21, A12 = 1.6f * b12 + 64.f * b22;

    float x = A00 * (float)w + A01 * (float)h + A02;
    float y = A10 * (float)w + A11 * (float)h + A12;

    float x0f = floorf(x), y0f = floorf(y);
    float fx = x - x0f, fy = y - y0f;
    float w00 = (1.f - fx) * (1.f - fy);
    float w10 = fx * (1.f - fy);
    float w01 = (1.f - fx) * fy;
    float w11 = fx * fy;
    float vx0 = (x0f >= 0.f        && x0f <= 127.f)        ? 1.f : 0.f;
    float vx1 = (x0f + 1.f >= 0.f  && x0f + 1.f <= 127.f)  ? 1.f : 0.f;
    float vy0 = (y0f >= 0.f        && y0f <= 127.f)        ? 1.f : 0.f;
    float vy1 = (y0f + 1.f >= 0.f  && y0f + 1.f <= 127.f)  ? 1.f : 0.f;
    int xi0 = (int)fminf(fmaxf(x0f,        0.f), 127.f);
    int xi1 = (int)fminf(fmaxf(x0f + 1.f,  0.f), 127.f);
    int yi0 = (int)fminf(fmaxf(y0f,        0.f), 127.f);
    int yi1 = (int)fminf(fmaxf(y0f + 1.f,  0.f), 127.f);

    const float* img = bev + frame + (size_t)c * kHW;
    v = w00 * vx0 * vy0 * img[yi0 * kW + xi0]
      + w10 * vx1 * vy0 * img[yi0 * kW + xi1]
      + w01 * vx0 * vy1 * img[yi1 * kW + xi0]
      + w11 * vx1 * vy1 * img[yi1 * kW + xi1];
  }
  X[((size_t)t * kHW + n) * kC + c] = (bf16_t)v;
}

// Assemble a 16-element bf16 WMMA fragment from two contiguous 16B loads.
__device__ __forceinline__ v16bf load_frag16(const bf16_t* p0, const bf16_t* p1) {
  v8bf lo = *(const v8bf*)p0;
  v8bf hi = *(const v8bf*)p1;
  v16bf r;
#pragma unroll
  for (int i = 0; i < 8; ++i) { r[i] = lo[i]; r[i + 8] = hi[i]; }
  return r;
}

__device__ __forceinline__ v8f wmma_bf16(v16bf a, v16bf b, v8f c) {
  return __builtin_amdgcn_wmma_f32_16x16x32_bf16(
      /*neg_a=*/false, a, /*neg_b=*/false, b,
      /*c_mod=*/(short)0, c, /*reuse_a=*/false, /*reuse_b=*/false);
}

// ---------------------------------------------------------------------------
// Conv1d as 3 accumulated GEMMs: Y[o][t][n] = sum_k Wp[k][o][:] . X[t+k-1][n][:]
// Block = 256 threads = 8 waves covering all 256 output channels of one
// 32-wide n-strip. B (activations) for each tap is staged into LDS once per
// block (async-to-LDS when available), then each wave runs a 2x2 register-
// blocked tile: 4 A-loads (global) + 4 B-loads (LDS) feed 4 wmma per K-chunk.
// ---------------------------------------------------------------------------
__global__ void __launch_bounds__(256)
k_conv1d_wmma(const bf16_t* __restrict__ X,
              const bf16_t* __restrict__ Wp,
              float* __restrict__ Y) {
  __shared__ bf16_t Bsh[3][32][kLP];   // ~49.5 KB of the 320 KB WGP LDS

  const int tid  = threadIdx.x;
  const int lane = tid & 31;
  const int wave = tid >> 5;
  const int m    = lane & 15;
  const int half = lane >> 4;
  const int Ns   = blockIdx.x * 32;    // n-strip base
  const int t    = blockIdx.y;

  // ---- stage the three B tiles (16KB each) into LDS ----
#pragma unroll
  for (int k = 0; k < 3; ++k) {
    const int tp = t + k - 1;          // uniform per block
    if (tp < 0 || tp > 7) continue;
    const bf16_t* Xt = X + (size_t)tp * kHW * kC;
#pragma unroll
    for (int issue = 0; issue < 4; ++issue) {
      int i   = issue * 256 + tid;     // 1024 chunks of 16B
      int row = i >> 5;                // 0..31
      int col = (i & 31) * 8;          // bf16 offset, 16B granules
      stage16(Xt + (size_t)(Ns + row) * kC + col, &Bsh[k][row][col]);
    }
  }
  stage_fence();

  // ---- 2x2 register-blocked WMMA ----
  const int o0 = wave * 32;            // this wave: o in [o0, o0+32)
  v8f acc00 = {}, acc01 = {}, acc10 = {}, acc11 = {};

#pragma unroll
  for (int k = 0; k < 3; ++k) {
    const int tp = t + k - 1;
    if (tp < 0 || tp > 7) continue;
    const bf16_t* a0row = Wp + (size_t)k * (kC * kC) + (size_t)(o0 + m) * kC + half * 8;
    const bf16_t* a1row = a0row + 16 * kC;
    const bf16_t* b0row = &Bsh[k][m][half * 8];
    const bf16_t* b1row = &Bsh[k][16 + m][half * 8];
#pragma unroll
    for (int K0 = 0; K0 < kC; K0 += 32) {
      if (K0 + 32 < kC) {
        __builtin_prefetch(a0row + K0 + 32, 0, 1);
        __builtin_prefetch(a1row + K0 + 32, 0, 1);
      }
      v16bf a0 = load_frag16(a0row + K0, a0row + K0 + 16);  // A: K 0-7 / 16-23
      v16bf a1 = load_frag16(a1row + K0, a1row + K0 + 16);
      v16bf b0 = load_frag16(b0row + K0, b0row + K0 + 16);  // B: K 0-7 / 16-23
      v16bf b1 = load_frag16(b1row + K0, b1row + K0 + 16);
      acc00 = wmma_bf16(a0, b0, acc00);
      acc01 = wmma_bf16(a0, b1, acc01);
      acc10 = wmma_bf16(a1, b0, acc10);
      acc11 = wmma_bf16(a1, b1, acc11);
    }
  }

#pragma unroll
  for (int r = 0; r < 8; ++r) {
    int oA = o0 + half * 8 + r;        // C/D layout: VGPR r -> M = r + 8*half
    int oB = oA + 16;
    Y[((size_t)oA * kT + t) * kHW + Ns + m]      = acc00[r];
    Y[((size_t)oA * kT + t) * kHW + Ns + 16 + m] = acc01[r];
    Y[((size_t)oB * kT + t) * kHW + Ns + m]      = acc10[r];
    Y[((size_t)oB * kT + t) * kHW + Ns + 16 + m] = acc11[r];
  }
}

// ---------------------------------------------------------------------------
// BatchNorm1d (batch stats over N*L=131072) + exact GELU.
// One block per channel o. Writes tokens output [n][c][t] and the bf16
// t=7 slice G[n][c] consumed by the projection GEMM.
// ---------------------------------------------------------------------------
__global__ void k_bn1_gelu(const float* __restrict__ Y,
                           const float* __restrict__ gamma,
                           const float* __restrict__ beta,
                           float* __restrict__ tokens_out,
                           bf16_t* __restrict__ G) {
  const int o = blockIdx.x;
  const int tid = threadIdx.x;
  const float* Yo = Y + (size_t)o * kNL;

  float s = 0.f, s2 = 0.f;
  for (int i = tid; i < kNL; i += 256) {
    float v = Yo[i];
    s += v; s2 += v * v;
  }
  __shared__ float sh[256], sh2[256];
  sh[tid] = s; sh2[tid] = s2;
  __syncthreads();
  for (int off = 128; off > 0; off >>= 1) {
    if (tid < off) { sh[tid] += sh[tid + off]; sh2[tid] += sh2[tid + off]; }
    __syncthreads();
  }
  const float inv = 1.0f / (float)kNL;
  float mean = sh[0] * inv;
  float var  = sh2[0] * inv - mean * mean;     // biased var (jnp.var default)
  float sc   = gamma[o] * rsqrtf(var + 1e-5f);
  float sb   = beta[o] - mean * sc;

  for (int i = tid; i < kNL; i += 256) {
    float v = Yo[i] * sc + sb;
    float g = 0.5f * v * (1.0f + erff(v * 0.70710678118654752f));
    int t = i >> 14;           // i / 16384
    int n = i & 16383;
    tokens_out[((size_t)n * kC + o) * kT + t] = g;
    if (t == 7) G[(size_t)n * kC + o] = (bf16_t)g;
  }
}

// ---------------------------------------------------------------------------
// 1x1 projection GEMM: F[o][n] = Pb[o][:] . G[n][:], K = 256.
// Same LDS-staged, 2x2 register-blocked scheme as the conv GEMM (one tap).
// ---------------------------------------------------------------------------
__global__ void __launch_bounds__(256)
k_proj_wmma(const bf16_t* __restrict__ G,
            const bf16_t* __restrict__ Pb,
            float* __restrict__ F) {
  __shared__ bf16_t Bsh[32][kLP];

  const int tid  = threadIdx.x;
  const int lane = tid & 31;
  const int wave = tid >> 5;
  const int m    = lane & 15;
  const int half = lane >> 4;
  const int Ns   = blockIdx.x * 32;

#pragma unroll
  for (int issue = 0; issue < 4; ++issue) {
    int i   = issue * 256 + tid;
    int row = i >> 5;
    int col = (i & 31) * 8;
    stage16(G + (size_t)(Ns + row) * kC + col, &Bsh[row][col]);
  }
  stage_fence();

  const int o0 = wave * 32;
  v8f acc00 = {}, acc01 = {}, acc10 = {}, acc11 = {};
  const bf16_t* a0row = Pb + (size_t)(o0 + m) * kC + half * 8;
  const bf16_t* a1row = a0row + 16 * kC;
  const bf16_t* b0row = &Bsh[m][half * 8];
  const bf16_t* b1row = &Bsh[16 + m][half * 8];

#pragma unroll
  for (int K0 = 0; K0 < kC; K0 += 32) {
    if (K0 + 32 < kC) {
      __builtin_prefetch(a0row + K0 + 32, 0, 1);
      __builtin_prefetch(a1row + K0 + 32, 0, 1);
    }
    v16bf a0 = load_frag16(a0row + K0, a0row + K0 + 16);
    v16bf a1 = load_frag16(a1row + K0, a1row + K0 + 16);
    v16bf b0 = load_frag16(b0row + K0, b0row + K0 + 16);
    v16bf b1 = load_frag16(b1row + K0, b1row + K0 + 16);
    acc00 = wmma_bf16(a0, b0, acc00);
    acc01 = wmma_bf16(a0, b1, acc01);
    acc10 = wmma_bf16(a1, b0, acc10);
    acc11 = wmma_bf16(a1, b1, acc11);
  }

#pragma unroll
  for (int r = 0; r < 8; ++r) {
    int oA = o0 + half * 8 + r;
    int oB = oA + 16;
    F[(size_t)oA * kHW + Ns + m]      = acc00[r];
    F[(size_t)oA * kHW + Ns + 16 + m] = acc01[r];
    F[(size_t)oB * kHW + Ns + m]      = acc10[r];
    F[(size_t)oB * kHW + Ns + 16 + m] = acc11[r];
  }
}

// ---------------------------------------------------------------------------
// BatchNorm2d (stats over H*W=16384) + exact GELU -> fused output [o][n].
// ---------------------------------------------------------------------------
__global__ void k_bn2_gelu(const float* __restrict__ F,
                           const float* __restrict__ gamma,
                           const float* __restrict__ beta,
                           float* __restrict__ fused_out) {
  const int o = blockIdx.x;
  const int tid = threadIdx.x;
  const float* Fo = F + (size_t)o * kHW;

  float s = 0.f, s2 = 0.f;
  for (int i = tid; i < kHW; i += 256) {
    float v = Fo[i];
    s += v; s2 += v * v;
  }
  __shared__ float sh[256], sh2[256];
  sh[tid] = s; sh2[tid] = s2;
  __syncthreads();
  for (int off = 128; off > 0; off >>= 1) {
    if (tid < off) { sh[tid] += sh[tid + off]; sh2[tid] += sh2[tid + off]; }
    __syncthreads();
  }
  const float inv = 1.0f / (float)kHW;
  float mean = sh[0] * inv;
  float var  = sh2[0] * inv - mean * mean;
  float sc   = gamma[o] * rsqrtf(var + 1e-5f);
  float sb   = beta[o] - mean * sc;

  for (int i = tid; i < kHW; i += 256) {
    float v = Fo[i] * sc + sb;
    fused_out[(size_t)o * kHW + i] = 0.5f * v * (1.0f + erff(v * 0.70710678118654752f));
  }
}

// ---------------------------------------------------------------------------
// Launch: inputs in setup_inputs() order.
// Workspace layout (bytes):
//   [0, 67108864)              X  bf16 [8][16384][256]
//   [67108864, 67502080)       Wp bf16 [3][256][256]
//   [67502080, 67633152)       Pb bf16 [256][256]
//   [67633152, 201850880)      Y  f32  [256][8][16384]
//   overlap (X dead after conv GEMM):
//   [0, 16777216)              F  f32  [256][16384]
//   [16777216, 25165824)       G  bf16 [16384][256]
// Total required: ~202 MB.
// ---------------------------------------------------------------------------
extern "C" void kernel_launch(void* const* d_in, const int* in_sizes, int n_in,
                              void* d_out, int out_size, void* d_ws, size_t ws_size,
                              hipStream_t stream) {
  const float* bev    = (const float*)d_in[0];
  const float* h2k    = (const float*)d_in[1];
  const float* conv_w = (const float*)d_in[2];
  const float* g1     = (const float*)d_in[3];
  const float* b1     = (const float*)d_in[4];
  const float* proj_w = (const float*)d_in[5];
  const float* g2     = (const float*)d_in[6];
  const float* b2     = (const float*)d_in[7];

  float* out        = (float*)d_out;
  float* fused_out  = out;                 // [1,256,128,128] = 4,194,304 floats
  float* tokens_out = out + 4194304;       // [16384,256,8]   = 33,554,432 floats

  char* ws = (char*)d_ws;
  bf16_t* X  = (bf16_t*)ws;
  bf16_t* Wp = (bf16_t*)(ws + 67108864);
  bf16_t* Pb = (bf16_t*)(ws + 67502080);
  float*  Y  = (float*) (ws + 67633152);
  float*  F  = (float*) ws;                // overlaps X (dead by then)
  bf16_t* G  = (bf16_t*)(ws + 16777216);   // overlaps X tail (dead by then)

  // 1. repack weights to bf16, GEMM-friendly layout
  k_pack_w<<<1024, 256, 0, stream>>>(conv_w, proj_w, Wp, Pb);
  // 2. warp history + pack activations to bf16 [t][n][c]
  k_align_pack<<<dim3(128, 128, 8), 256, 0, stream>>>(bev, h2k, X);
  // 3. temporal conv as LDS-staged, register-blocked WMMA GEMMs
  k_conv1d_wmma<<<dim3(512, 8), 256, 0, stream>>>(X, Wp, Y);
  // 4. BN1 + GELU -> tokens output + bf16 last-slice
  k_bn1_gelu<<<256, 256, 0, stream>>>(Y, g1, b1, tokens_out, G);
  // 5. 1x1 projection via WMMA
  k_proj_wmma<<<512, 256, 0, stream>>>(G, Pb, F);
  // 6. BN2 + GELU -> fused output
  k_bn2_gelu<<<256, 256, 0, stream>>>(F, g2, b2, fused_out);
}
